// SWA_48249662603691
// MI455X (gfx1250) — compile-verified, matrix operand order checked by reading, not run on hardware
//
#include <hip/hip_runtime.h>
#include <hip/hip_bf16.h>

// ---------------- problem constants (fixed shapes from the reference) -------
#define BB   2
#define SS   2048
#define EMB  2048
#define NH   16
#define NKV  8
#define HD   128
#define KV   1024          // NKV * HD
#define MT   (BB * SS)     // 4096 token rows

typedef __bf16 bf16_t;
typedef __attribute__((ext_vector_type(8)))  bf16_t v8bf;
typedef __attribute__((ext_vector_type(16))) bf16_t v16bf;
typedef __attribute__((ext_vector_type(8)))  float  v8f;

// ---------------- WMMA helpers ---------------------------------------------
__device__ __forceinline__ v8f wmma_bf16(v16bf a, v16bf b, v8f c) {
    // D = A(16x32 bf16) * B(32x16 bf16) + C(16x16 f32)
    return __builtin_amdgcn_wmma_f32_16x16x32_bf16(
        /*neg_a=*/false, a, /*neg_b=*/false, b,
        /*c_mod=*/(short)0, c, /*reuse_a=*/false, /*reuse_b=*/false);
}

__device__ __forceinline__ v16bf mk16(v8bf lo, v8bf hi) {
    v16bf r;
#pragma unroll
    for (int i = 0; i < 8; ++i) { r[i] = lo[i]; r[i + 8] = hi[i]; }
    return r;
}

// Load a 16x32 bf16 fragment (A or B layout, both row-of-K-contiguous):
// p points at element (row=lane%16, k = kbase + (lane/16)*8); fetch 8 halves
// there and 8 halves at +16 (per ISA 7.12.2 16-bit layout).
__device__ __forceinline__ v16bf load_frag16(const bf16_t* p) {
    v8bf lo = *reinterpret_cast<const v8bf*>(p);
    v8bf hi = *reinterpret_cast<const v8bf*>(p + 16);
    return mk16(lo, hi);
}

__device__ __forceinline__ float redmax16(float v) {
#pragma unroll
    for (int m = 1; m < 16; m <<= 1) v = fmaxf(v, __shfl_xor(v, m, 32));
    return v;
}
__device__ __forceinline__ float redsum16(float v) {
#pragma unroll
    for (int m = 1; m < 16; m <<= 1) v += __shfl_xor(v, m, 32);
    return v;
}

// ---------------- CDNA5 async copy to LDS (ASYNCcnt path) ------------------
// Moves 16 bytes per lane from global memory straight into LDS without
// touching VGPRs; tracked by ASYNCcnt. LDS byte address = addr[31:0] of the
// generic shared pointer (ISA 10.2: LDS aperture truncation).
__device__ __forceinline__ void async_cp16(const bf16_t* g, unsigned lds_off) {
    unsigned long long ga = (unsigned long long)(uintptr_t)g;
    asm volatile("global_load_async_to_lds_b128 %0, %1, off"
                 :: "v"(lds_off), "v"(ga) : "memory");
}
__device__ __forceinline__ void wait_async0() {
    asm volatile("s_wait_asynccnt 0x0" ::: "memory");
}
__device__ __forceinline__ unsigned lds_addr32(const void* p) {
    return (unsigned)(uintptr_t)p;   // low 32 bits = LDS byte offset
}

// ---------------- fp32 -> bf16 cast ----------------------------------------
__global__ void swa_cast_kernel(const float* __restrict__ in,
                                bf16_t* __restrict__ out, int n) {
    int i = blockIdx.x * blockDim.x + threadIdx.x;
    if (i < n) out[i] = (bf16_t)in[i];
}

// ---------------- GEMM: OUT[M,N] = A[M,K](bf16) @ Bw[N,K]^T (bf16) + bias ---
// 256 threads = 8 waves; block tile 128x64; wave tile 32x32 (2x2 WMMA tiles).
// A/B tiles staged into LDS with async-to-LDS DMA, double buffered so the
// async fill of tile k+1 overlaps WMMAs on tile k.
__global__ __launch_bounds__(256)
void swa_gemm_kernel(const bf16_t* __restrict__ A, const bf16_t* __restrict__ Bw,
                     const float* __restrict__ bias,
                     float* __restrict__ outF, bf16_t* __restrict__ outB,
                     int M, int N, int K) {
    __shared__ __attribute__((aligned(16))) bf16_t lds_a[2][128][32]; // 16 KB
    __shared__ __attribute__((aligned(16))) bf16_t lds_b[2][64][32];  //  8 KB

    const int t    = threadIdx.x;
    const int w    = t >> 5;
    const int lane = t & 31;
    const int half = lane >> 4;
    const int lr   = lane & 15;

    const int m0 = blockIdx.y * 128 + (w & 3) * 32;
    const int n0 = blockIdx.x * 64  + (w >> 2) * 32;

    // copy assignment: thread t moves rows (t/4, t/4+64) of A-tile and row t/4
    // of B-tile, 16B (8 bf16) at column quarter (t%4)*8.
    const int rowC = t >> 2;
    const int colC = (t & 3) * 8;
    const bf16_t* gA0 = A  + (size_t)(blockIdx.y * 128 + rowC)      * K + colC;
    const bf16_t* gA1 = A  + (size_t)(blockIdx.y * 128 + rowC + 64) * K + colC;
    const bf16_t* gB0 = Bw + (size_t)(blockIdx.x * 64  + rowC)      * K + colC;

    v8f z = {0.f,0.f,0.f,0.f,0.f,0.f,0.f,0.f};
    v8f acc00 = z, acc01 = z, acc10 = z, acc11 = z;

    // prologue: async-fill buffer 0 with the k=0 tiles
    async_cp16(gA0, lds_addr32(&lds_a[0][rowC][colC]));
    async_cp16(gA1, lds_addr32(&lds_a[0][rowC + 64][colC]));
    async_cp16(gB0, lds_addr32(&lds_b[0][rowC][colC]));

    int buf = 0;
    for (int kk = 0; kk < K; kk += 32) {
        wait_async0();          // this wave's fills of lds[buf] have landed
        __syncthreads();        // every wave's fills have landed / reads done
        const int nbuf = buf ^ 1;
        if (kk + 32 < K) {      // overlap: DMA next tile while computing
            async_cp16(gA0 + kk + 32, lds_addr32(&lds_a[nbuf][rowC][colC]));
            async_cp16(gA1 + kk + 32, lds_addr32(&lds_a[nbuf][rowC + 64][colC]));
            async_cp16(gB0 + kk + 32, lds_addr32(&lds_b[nbuf][rowC][colC]));
        }
        const bf16_t* la = &lds_a[buf][(w & 3) * 32 + lr][half * 8];
        const bf16_t* lb = &lds_b[buf][(w >> 2) * 32 + lr][half * 8];
        v16bf a0 = load_frag16(la);
        v16bf a1 = load_frag16(la + 16 * 32);
        v16bf b0 = load_frag16(lb);
        v16bf b1 = load_frag16(lb + 16 * 32);
        acc00 = wmma_bf16(a0, b0, acc00);
        acc01 = wmma_bf16(a0, b1, acc01);
        acc10 = wmma_bf16(a1, b0, acc10);
        acc11 = wmma_bf16(a1, b1, acc11);
        buf = nbuf;
    }

    const float bv0 = bias ? bias[n0 + lr]      : 0.0f;
    const float bv1 = bias ? bias[n0 + 16 + lr] : 0.0f;
#pragma unroll
    for (int r = 0; r < 8; ++r) {
        const int mmA = m0 + half * 8 + r;       // C layout: M = r + half*8
        const int mmB = mmA + 16;
        const int nnA = n0 + lr;
        const int nnB = nnA + 16;
        float v00 = acc00[r] + bv0, v01 = acc01[r] + bv1;
        float v10 = acc10[r] + bv0, v11 = acc11[r] + bv1;
        if (outF) {
            outF[(size_t)mmA * N + nnA] = v00;
            outF[(size_t)mmA * N + nnB] = v01;
            outF[(size_t)mmB * N + nnA] = v10;
            outF[(size_t)mmB * N + nnB] = v11;
        } else {
            outB[(size_t)mmA * N + nnA] = (bf16_t)v00;
            outB[(size_t)mmA * N + nnB] = (bf16_t)v01;
            outB[(size_t)mmB * N + nnA] = (bf16_t)v10;
            outB[(size_t)mmB * N + nnB] = (bf16_t)v11;
        }
    }
}

// ---------------- interleaved RoPE (bf16 in/out, fp32 math) ----------------
__global__ void swa_rope_kernel(const bf16_t* __restrict__ in,
                                bf16_t* __restrict__ out, int rows, int dim) {
    int idx = blockIdx.x * blockDim.x + threadIdx.x;
    int pairs = dim >> 1;
    if (idx >= rows * pairs) return;
    int cp  = idx % pairs;
    int row = idx / pairs;
    int s   = row % SS;
    int p   = cp & 63;                         // pair index within head (HD=128)
    int col = ((cp >> 6) << 7) + (p << 1);     // head*128 + 2*p
    float inv = __powf(10000.0f, -(float)p / 64.0f);
    float ang = (float)s * inv;
    float sn, cs;
    __sincosf(ang, &sn, &cs);
    size_t base = (size_t)row * dim + col;
    float x1 = (float)in[base], x2 = (float)in[base + 1];
    out[base]     = (bf16_t)(x1 * cs - x2 * sn);
    out[base + 1] = (bf16_t)(x1 * sn + x2 * cs);
}

// ---------------- V transpose: [B,S,KV] -> [B,KV,S] ------------------------
__global__ void swa_vtrans_kernel(const bf16_t* __restrict__ in,
                                  bf16_t* __restrict__ out) {
    int idx = blockIdx.x * blockDim.x + threadIdx.x;
    if (idx >= BB * SS * KV) return;
    int d = idx % KV;
    int s = (idx / KV) % SS;
    int b = idx / (KV * SS);
    out[((size_t)b * KV + d) * SS + s] = in[idx];
}

// ---------------- flash-style sliding-window attention ----------------------
// grid = B*NH*(S/128) blocks, 256 thr = 8 waves; each wave owns 16 q rows.
__global__ __launch_bounds__(256)
void swa_attn_kernel(const bf16_t* __restrict__ q, const bf16_t* __restrict__ k,
                     const bf16_t* __restrict__ vt, bf16_t* __restrict__ ctx,
                     const int* __restrict__ winp) {
    __shared__ __attribute__((aligned(16))) bf16_t lds_p[8][16][32];

    const int window = *winp;
    const int blk  = blockIdx.x;
    const int qblk = blk & 15;          // S/128 = 16
    const int h    = (blk >> 4) & 15;
    const int b    = blk >> 8;
    const int w    = threadIdx.x >> 5;
    const int lane = threadIdx.x & 31;
    const int half = lane >> 4;
    const int lr   = lane & 15;

    const int q0  = qblk * 128 + w * 16;
    const int kvh = h >> 1;             // GQA: 2 q-heads per kv-head

    // Q fragments: 4 k-tiles covering HEAD_DIM=128
    v16bf qf[4];
#pragma unroll
    for (int dt = 0; dt < 4; ++dt) {
        const bf16_t* qp = q + (size_t)(b * SS + q0 + lr) * EMB
                             + h * HD + dt * 32 + half * 8;
        qf[dt] = load_frag16(qp);
    }

    v8f zero = {0.f,0.f,0.f,0.f,0.f,0.f,0.f,0.f};
    v8f acc[8];
    float m[8], l[8];
#pragma unroll
    for (int nt = 0; nt < 8; ++nt) acc[nt] = zero;
#pragma unroll
    for (int r = 0; r < 8; ++r) { m[r] = -1e30f; l[r] = 0.0f; }

    int lo = q0 - window; if (lo < 0) lo = 0;
    const int kstart = lo & ~31;
    const int kend   = q0 + 16;
    const float scale = 0.08838834764831845f;  // 1/sqrt(128)

    for (int kb = kstart; kb < kend; kb += 32) {
        // ---- scores: two 16x16 tiles (keys kb..kb+15, kb+16..kb+31) -------
        v8f s0 = zero, s1 = zero;
#pragma unroll
        for (int dt = 0; dt < 4; ++dt) {
            const bf16_t* kp0 = k + (size_t)(b * SS + kb + lr) * KV
                                  + kvh * HD + dt * 32 + half * 8;
            s0 = wmma_bf16(qf[dt], load_frag16(kp0), s0);
            const bf16_t* kp1 = kp0 + (size_t)16 * KV;
            s1 = wmma_bf16(qf[dt], load_frag16(kp1), s1);
        }
        // ---- scale + sliding-window causal mask ---------------------------
#pragma unroll
        for (int r = 0; r < 8; ++r) {
            const int i  = q0 + half * 8 + r;
            const int j0 = kb + lr;
            const int j1 = j0 + 16;
            float a = s0[r] * scale, c = s1[r] * scale;
            s0[r] = ((j0 <= i) && (i - j0 <= window)) ? a : -1e30f;
            s1[r] = ((j1 <= i) && (i - j1 <= window)) ? c : -1e30f;
        }
        // ---- online softmax update ---------------------------------------
        float f[8];
#pragma unroll
        for (int r = 0; r < 8; ++r) {
            float t  = redmax16(fmaxf(s0[r], s1[r]));
            float mn = fmaxf(m[r], t);
            f[r] = __expf(m[r] - mn);
            float p0 = __expf(s0[r] - mn);
            float p1 = __expf(s1[r] - mn);
            lds_p[w][half * 8 + r][lr]      = (bf16_t)p0;
            lds_p[w][half * 8 + r][16 + lr] = (bf16_t)p1;
            l[r] = l[r] * f[r] + redsum16(p0 + p1);
            m[r] = mn;
        }
#pragma unroll
        for (int nt = 0; nt < 8; ++nt)
#pragma unroll
            for (int r = 0; r < 8; ++r) acc[nt][r] *= f[r];

        // ---- P (C-layout) -> A-layout via LDS, then PV on WMMA ------------
        v8bf plo = *reinterpret_cast<const v8bf*>(&lds_p[w][lr][half * 8]);
        v8bf phi = *reinterpret_cast<const v8bf*>(&lds_p[w][lr][half * 8 + 16]);
        v16bf pf = mk16(plo, phi);
#pragma unroll
        for (int nt = 0; nt < 8; ++nt) {
            const bf16_t* vp = vt + ((size_t)b * KV + kvh * HD + nt * 16 + lr) * SS
                                  + kb + half * 8;
            acc[nt] = wmma_bf16(pf, load_frag16(vp), acc[nt]);
        }
    }

    // ---- normalize + store context bf16 -----------------------------------
#pragma unroll
    for (int r = 0; r < 8; ++r) {
        const float invl = 1.0f / l[r];
        const int row = q0 + half * 8 + r;
#pragma unroll
        for (int nt = 0; nt < 8; ++nt) {
            ctx[(size_t)(b * SS + row) * EMB + h * HD + nt * 16 + lr] =
                (bf16_t)(acc[nt][r] * invl);
        }
    }
}

// ---------------- host side -------------------------------------------------
extern "C" void kernel_launch(void* const* d_in, const int* in_sizes, int n_in,
                              void* d_out, int out_size, void* d_ws, size_t ws_size,
                              hipStream_t stream) {
    const float* x    = (const float*)d_in[0];
    const float* wq_w = (const float*)d_in[1];
    const float* wq_b = (const float*)d_in[2];
    const float* wk_w = (const float*)d_in[3];
    const float* wk_b = (const float*)d_in[4];
    const float* wv_w = (const float*)d_in[5];
    const float* wv_b = (const float*)d_in[6];
    const float* wo_w = (const float*)d_in[7];
    const int*   win  = (const int*)  d_in[8];

    // workspace layout (bytes)
    char* p = (char*)d_ws;
    const size_t MB = 1024 * 1024;
    bf16_t* xb   = (bf16_t*)(p +   0 * MB);   // 16 MB  x bf16
    bf16_t* wqb  = (bf16_t*)(p +  16 * MB);   //  8 MB
    bf16_t* wkb  = (bf16_t*)(p +  24 * MB);   //  4 MB
    bf16_t* wvb  = (bf16_t*)(p +  28 * MB);   //  4 MB
    bf16_t* wob  = (bf16_t*)(p +  32 * MB);   //  8 MB
    bf16_t* qpre = (bf16_t*)(p +  40 * MB);   // 16 MB  pre-RoPE Q
    bf16_t* kpre = (bf16_t*)(p +  56 * MB);   //  8 MB  pre-RoPE K
    bf16_t* vpre = (bf16_t*)(p +  64 * MB);   //  8 MB  V
    bf16_t* qr   = (bf16_t*)(p +  72 * MB);   // 16 MB  RoPE'd Q
    bf16_t* kr   = (bf16_t*)(p +  88 * MB);   //  8 MB  RoPE'd K
    bf16_t* vtb  = (bf16_t*)(p +  96 * MB);   //  8 MB  V transposed [B,KV,S]
    bf16_t* ctxb = (bf16_t*)(p + 104 * MB);   // 16 MB  attention context

    const int T = 256;
    auto grid1 = [](int n, int t) { return dim3((n + t - 1) / t); };

    // 1) cast inputs/weights to bf16
    swa_cast_kernel<<<grid1(MT * EMB, T), T, 0, stream>>>(x, xb, MT * EMB);
    swa_cast_kernel<<<grid1(EMB * EMB, T), T, 0, stream>>>(wq_w, wqb, EMB * EMB);
    swa_cast_kernel<<<grid1(KV * EMB, T), T, 0, stream>>>(wk_w, wkb, KV * EMB);
    swa_cast_kernel<<<grid1(KV * EMB, T), T, 0, stream>>>(wv_w, wvb, KV * EMB);
    swa_cast_kernel<<<grid1(EMB * EMB, T), T, 0, stream>>>(wo_w, wob, EMB * EMB);

    // 2) QKV projections (bf16 WMMA, async-LDS staged, bf16 out)
    swa_gemm_kernel<<<dim3(EMB / 64, MT / 128), T, 0, stream>>>(
        xb, wqb, wq_b, nullptr, qpre, MT, EMB, EMB);
    swa_gemm_kernel<<<dim3(KV / 64, MT / 128), T, 0, stream>>>(
        xb, wkb, wk_b, nullptr, kpre, MT, KV, EMB);
    swa_gemm_kernel<<<dim3(KV / 64, MT / 128), T, 0, stream>>>(
        xb, wvb, wv_b, nullptr, vpre, MT, KV, EMB);

    // 3) RoPE on Q, K; transpose V for K-contiguous PV GEMM
    swa_rope_kernel<<<grid1(MT * (EMB / 2), T), T, 0, stream>>>(qpre, qr, MT, EMB);
    swa_rope_kernel<<<grid1(MT * (KV / 2), T), T, 0, stream>>>(kpre, kr, MT, KV);
    swa_vtrans_kernel<<<grid1(BB * SS * KV, T), T, 0, stream>>>(vpre, vtb);

    // 4) flash-style sliding-window attention (WMMA QK^T and PV)
    swa_attn_kernel<<<dim3(BB * NH * (SS / 128)), T, 0, stream>>>(
        qr, kr, vtb, ctxb, win);

    // 5) output projection -> fp32 d_out
    swa_gemm_kernel<<<dim3(EMB / 64, MT / 128), T, 0, stream>>>(
        ctxb, wob, nullptr, (float*)d_out, nullptr, MT, EMB, EMB);
}
